// meshProteinFeaturesold_11115375362500
// MI455X (gfx1250) — compile-verified, matrix operand order checked by reading, not run on hardware
//
#include <hip/hip_runtime.h>
#include <hip/hip_bf16.h>
#include <math.h>

// ---------------- problem constants ----------------
#define BB     8
#define LL     1024
#define KNB    30
#define NUM_RBF 16
#define NPOS   16
#define MAXREL 32
#define EDGE   128
#define EDGE_IN (NPOS + NUM_RBF * 25)   // 416
#define KPAD   448                      // 14 * 32 (pad K for WMMA)
#define NKT    14                       // K tiles of 32
#define NNT    8                        // N tiles of 16 (128/16)
#define NEDGE  (BB * LL * KNB)          // 245760
#define NTILE  (NEDGE / 16)             // 15360 (exact)

typedef __attribute__((ext_vector_type(16))) _Float16 v16h;
typedef __attribute__((ext_vector_type(8)))  _Float16 v8h;
typedef __attribute__((ext_vector_type(8)))  float    v8f;

__device__ __constant__ int c_PI[25] = {0,1,2,3,4,0,0,0,0,1,1,1,4,4,3,1,2,3,4,2,3,4,2,3,2};
__device__ __constant__ int c_PJ[25] = {0,1,2,3,4,1,2,3,4,2,3,4,2,3,2,0,0,0,0,1,1,1,4,4,3};

// ---------------- kernel 1: atoms = [Ca, N, C, O, Cb] per residue ----------------
__global__ void k_atoms(const float* __restrict__ X, float* __restrict__ atoms) {
    int idx = blockIdx.x * blockDim.x + threadIdx.x;
    if (idx >= BB * LL) return;
    const float* x = X + (size_t)idx * 12; // (4 atoms, 3)
    float Nx = x[0], Ny = x[1],  Nz = x[2];
    float Cax= x[3], Cay= x[4],  Caz= x[5];
    float Cx = x[6], Cy = x[7],  Cz = x[8];
    float Ox = x[9], Oy = x[10], Oz = x[11];
    float bx = Cax - Nx, by = Cay - Ny, bz = Caz - Nz;
    float cx = Cx - Cax, cy = Cy - Cay, cz = Cz - Caz;
    float ax = by * cz - bz * cy;
    float ay = bz * cx - bx * cz;
    float az = bx * cy - by * cx;
    float Cbx = -0.58273431f * ax + 0.56802827f * bx - 0.54067466f * cx + Cax;
    float Cby = -0.58273431f * ay + 0.56802827f * by - 0.54067466f * cy + Cay;
    float Cbz = -0.58273431f * az + 0.56802827f * bz - 0.54067466f * cz + Caz;
    float* o = atoms + (size_t)idx * 15;
    o[0]=Cax; o[1]=Cay; o[2]=Caz;   // 0: Ca
    o[3]=Nx;  o[4]=Ny;  o[5]=Nz;    // 1: N
    o[6]=Cx;  o[7]=Cy;  o[8]=Cz;    // 2: C
    o[9]=Ox;  o[10]=Oy; o[11]=Oz;   // 3: O
    o[12]=Cbx;o[13]=Cby;o[14]=Cbz;  // 4: Cb
}

// ---------------- kernel 2: pre-swizzle W_edge into WMMA B-fragment layout ----
// Wt[nt][kt][lane][h] : 16-bit B 32x16 layout; lane = N%16 + 16*half,
// halves h: 0..7 -> K = kt*32 + half*8 + h ; 8..15 -> K = kt*32 + 16 + half*8 + (h-8)
__global__ void k_wtiles(const float* __restrict__ W_edge, _Float16* __restrict__ Wt) {
    int t = blockIdx.x * blockDim.x + threadIdx.x;
    if (t >= NNT * NKT * 32 * 16) return;
    int h    = t & 15;
    int lane = (t >> 4) & 31;
    int kt   = (t >> 9) % NKT;
    int nt   = (t >> 9) / NKT;
    int half = lane >> 4;
    int n    = nt * 16 + (lane & 15);
    int grp  = h >> 3, within = h & 7;
    int k    = kt * 32 + grp * 16 + half * 8 + within;
    float v  = (k < EDGE_IN) ? W_edge[(size_t)k * EDGE + n] : 0.0f;
    Wt[t] = (_Float16)v;
}

// ---------------- kernel 3: masked kNN (top-K smallest D_adjust) ------------
__global__ void __launch_bounds__(256) k_topk(const float* __restrict__ atoms,
                                              const float* __restrict__ mask,
                                              int* __restrict__ Eidx) {
    __shared__ float Dadj[LL];                 // 4 KB
    __shared__ float fred[256];
    __shared__ unsigned long long red[256];
    int bi  = blockIdx.x;                      // flat (b,i)
    int b   = bi / LL;
    int tid = threadIdx.x;
    const float* ai = atoms + (size_t)bi * 15;
    float cax = ai[0], cay = ai[1], caz = ai[2];
    float mi  = mask[bi];
    float lmax = 0.0f;
    for (int j = tid; j < LL; j += 256) {
        const float* aj = atoms + ((size_t)b * LL + j) * 15;
        float dx = cax - aj[0], dy = cay - aj[1], dz = caz - aj[2];
        float m2 = mi * mask[b * LL + j];
        float D  = m2 * sqrtf(dx * dx + dy * dy + dz * dz + 1e-6f);
        Dadj[j]  = D;
        lmax     = fmaxf(lmax, D);
    }
    fred[tid] = lmax;
    __syncthreads();
    for (int s = 128; s > 0; s >>= 1) {
        if (tid < s) fred[tid] = fmaxf(fred[tid], fred[tid + s]);
        __syncthreads();
    }
    float Dmax = fred[0];
    for (int j = tid; j < LL; j += 256) {
        float m2 = mi * mask[b * LL + j];
        Dadj[j] = Dadj[j] + (1.0f - m2) * Dmax;
    }
    __syncthreads();
    for (int k = 0; k < KNB; k++) {
        unsigned long long best = ~0ull;
        for (int j = tid; j < LL; j += 256) {
            unsigned fb = __float_as_uint(Dadj[j]);   // non-negative: uint order == float order
            unsigned long long key = ((unsigned long long)fb << 32) | (unsigned)j;
            best = key < best ? key : best;
        }
        red[tid] = best;
        __syncthreads();
        for (int s = 128; s > 0; s >>= 1) {
            if (tid < s && red[tid + s] < red[tid]) red[tid] = red[tid + s];
            __syncthreads();
        }
        int jwin = (int)(red[0] & 0xffffffffu);
        if (tid == 0) {
            Eidx[(size_t)bi * KNB + k] = jwin;
            Dadj[jwin] = 3.0e38f;
        }
        __syncthreads();
    }
}

// ---------------- kernel 4: fused features + WMMA GEMM + LayerNorm ----------
// block = 256 threads = 8 waves; one block per 16 consecutive edges.
__global__ void __launch_bounds__(256) k_edge(const float* __restrict__ atoms,
                                              const int*   __restrict__ Eidx,
                                              const int*   __restrict__ residx,
                                              const float* __restrict__ W_pos,
                                              const float* __restrict__ b_pos,
                                              const _Float16* __restrict__ Wt,
                                              const float* __restrict__ ln_g,
                                              const float* __restrict__ ln_b,
                                              float* __restrict__ Eout) {
    __shared__ __align__(32) _Float16 Afeat[16][KPAD]; // 14 KB: 16 edges x 448 feats
    __shared__ float Etile[16][EDGE];                  // 8 KB
    int tid = threadIdx.x;
    int e0  = blockIdx.x * 16;

    // ---- feature build: 16 edges x 28 chunks of 16 halves (chunks 26,27 = pad) ----
    for (int task = tid; task < 16 * 28; task += 256) {
        int el = task / 28, ch = task % 28;
        int e  = e0 + el;
        int b  = e / (KNB * LL);
        int ii = (e / KNB) % LL;
        int j  = Eidx[e];
        float vals[16];
        if (ch == 0) {
            int off = residx[b * LL + ii] - residx[b * LL + j];
            int d   = off + MAXREL;
            d = d < 0 ? 0 : (d > 2 * MAXREL ? 2 * MAXREL : d);
            for (int r = 0; r < 16; r++) vals[r] = W_pos[d * NPOS + r] + b_pos[r];
        } else if (ch <= 25) {
            int p  = ch - 1;
            const float* ap = atoms + ((size_t)b * LL + ii) * 15 + c_PI[p] * 3;
            const float* aq = atoms + ((size_t)b * LL + j)  * 15 + c_PJ[p] * 3;
            float dx = ap[0] - aq[0], dy = ap[1] - aq[1], dz = ap[2] - aq[2];
            float dist = sqrtf(dx * dx + dy * dy + dz * dz + 1e-6f);
            for (int r = 0; r < 16; r++) {
                float mu = 2.0f + (float)r * (20.0f / 15.0f);
                float t  = (dist - mu) * (1.0f / 1.25f);
                vals[r]  = __expf(-t * t);
            }
        } else {
            for (int r = 0; r < 16; r++) vals[r] = 0.0f;
        }
        for (int r = 0; r < 16; r++) Afeat[el][ch * 16 + r] = (_Float16)vals[r];
    }
    __syncthreads();

    // ---- WMMA: each wave owns one N-tile of 16 channels -----------------------
    int wave = tid >> 5, lane = tid & 31;
    int m = lane & 15, hlf = lane >> 4;
    v8f acc = {};
    const _Float16* arow = &Afeat[m][0];
    for (int kt = 0; kt < NKT; kt++) {
        union { v16h v; v8h h[2]; } af;
        af.h[0] = *(const v8h*)(arow + kt * 32 + hlf * 8);        // K = kt*32 + hlf*8 + 0..7
        af.h[1] = *(const v8h*)(arow + kt * 32 + 16 + hlf * 8);   // K = kt*32 + 16 + hlf*8 + 0..7
        v16h bm = *(const v16h*)(Wt + (((size_t)wave * NKT + kt) * 32 + lane) * 16);
        acc = __builtin_amdgcn_wmma_f32_16x16x32_f16(false, af.v, false, bm,
                                                     (short)0, acc, false, false);
    }
    // spill D tile: VGPR r -> row = r + 8*half, col = wave*16 + (lane&15)
    for (int r = 0; r < 8; r++) Etile[r + 8 * hlf][wave * 16 + m] = acc[r];
    __syncthreads();

    // ---- LayerNorm over 128 channels per edge (16 threads per edge) -----------
    int el  = tid >> 4;
    int c16 = tid & 15;
    float s = 0.0f, ss = 0.0f;
    for (int q = 0; q < 8; q++) {
        float v = Etile[el][c16 + q * 16];
        s += v; ss += v * v;
    }
    for (int off = 8; off >= 1; off >>= 1) {
        s  += __shfl_xor(s,  off, 16);
        ss += __shfl_xor(ss, off, 16);
    }
    float mean = s * (1.0f / 128.0f);
    float var  = ss * (1.0f / 128.0f) - mean * mean;
    float inv  = rsqrtf(var + 1e-5f);
    float* orow = Eout + (size_t)(e0 + el) * EDGE;
    for (int q = 0; q < 8; q++) {
        int c = c16 + q * 16;
        orow[c] = (Etile[el][c] - mean) * inv * ln_g[c] + ln_b[c];
    }
}

// ---------------- host launch -------------------------------------------------
extern "C" void kernel_launch(void* const* d_in, const int* in_sizes, int n_in,
                              void* d_out, int out_size, void* d_ws, size_t ws_size,
                              hipStream_t stream) {
    const float* X      = (const float*)d_in[0];
    const float* mask   = (const float*)d_in[1];
    const int*   residx = (const int*)  d_in[2];
    const float* W_pos  = (const float*)d_in[6];
    const float* b_pos  = (const float*)d_in[7];
    const float* W_edge = (const float*)d_in[8];
    const float* ln_g   = (const float*)d_in[9];
    const float* ln_b   = (const float*)d_in[10];

    float* Eout = (float*)d_out;
    int*   Eidx = (int*)d_out + (size_t)NEDGE * EDGE;   // second tuple output

    const size_t ATOMS_BYTES = (size_t)BB * LL * 15 * sizeof(float);   // 491,520
    float*     atoms = (float*)d_ws;
    _Float16*  Wt    = (_Float16*)((char*)d_ws + ATOMS_BYTES);         // 114,688 B

    k_atoms<<<(BB * LL + 255) / 256, 256, 0, stream>>>(X, atoms);
    k_wtiles<<<(NNT * NKT * 32 * 16 + 255) / 256, 256, 0, stream>>>(W_edge, Wt);
    k_topk<<<BB * LL, 256, 0, stream>>>(atoms, mask, Eidx);
    k_edge<<<NTILE, 256, 0, stream>>>(atoms, Eidx, residx, W_pos, b_pos, Wt,
                                      ln_g, ln_b, Eout);
}
